// GPM_38053410242894
// MI455X (gfx1250) — matrix operand order repaired and checked by reading, not compiled, not run on hardware
//
#include <hip/hip_runtime.h>
#include <math.h>

typedef __attribute__((ext_vector_type(16))) _Float16 v16h;
typedef __attribute__((ext_vector_type(8)))  _Float16 v8h;
typedef __attribute__((ext_vector_type(4)))  _Float16 v4h;
typedef __attribute__((ext_vector_type(8)))  float    v8f;

#define C_DIM    512
#define KSTEPS   (C_DIM / 32)     // 16 wmma k-steps of K=32
#define TOPK     5
#define WAVES    8                // 4 row-tiles x 2 column partitions
#define THREADS  (WAVES * 32)     // 256
#define ROWS_BLK 64               // 4 row-tiles of 16
#define CPARTS   2
#define CANDR    (CPARTS * 16 * TOPK)   // 160 candidates per row at merge

struct BTile { v16h f[KSTEPS]; };       // 128 VGPRs of B fragments

// ---------------------------------------------------------------------------
// Kernel 1: L2-normalize each 512-wide row of `src`, write fp16 row to `dst`.
// ---------------------------------------------------------------------------
__global__ __launch_bounds__(128)
void GPM_rownorm_f16(const float* __restrict__ src, _Float16* __restrict__ dst) {
    const int row = blockIdx.x;
    const int tid = threadIdx.x;
    const size_t base = (size_t)row * C_DIM + (size_t)tid * 4;

    const float4 v = *(const float4*)(src + base);
    float ss = v.x * v.x + v.y * v.y + v.z * v.z + v.w * v.w;

    __shared__ float red[128];
    red[tid] = ss;
    __syncthreads();
    #pragma unroll
    for (int s = 64; s > 0; s >>= 1) {
        if (tid < s) red[tid] += red[tid + s];
        __syncthreads();
    }
    const float scale = 1.0f / fmaxf(sqrtf(red[0]), 1e-12f);

    v4h o;
    o[0] = (_Float16)(v.x * scale);
    o[1] = (_Float16)(v.y * scale);
    o[2] = (_Float16)(v.z * scale);
    o[3] = (_Float16)(v.w * scale);
    *(v4h*)(dst + base) = o;
}

// ---------------------------------------------------------------------------
// Kernel 2: fused  sim = qn @ mn^T  ->  streaming top-5  ->  softmax ->
//           gather fp32 rows -> out = x + 0.5 * retrieved.
//
// Each wave owns one 16-row A tile (128 VGPRs); 4 waves per column partition
// sweep the same column-tile sequence (B dedups in WGP$, L2 traffic ~4 GB).
// B is double-buffered across tiles with sched_barrier(0) so next-tile loads
// overlap the current tile's 16 back-to-back v_wmma_f32_16x16x32_f16.
// ---------------------------------------------------------------------------
__global__ __launch_bounds__(THREADS)
void GPM_simtopk(const _Float16* __restrict__ qn,   // [BS, 512] fp16 normalized
                 const _Float16* __restrict__ mn,   // [M, 512]  fp16 normalized (L2-resident)
                 const float*    __restrict__ x,    // [BS, 512] fp32
                 const float*    __restrict__ mem,  // [M, 512]  fp32 raw table
                 float*          __restrict__ out,  // [BS, 512] fp32
                 int M) {
    const int tid     = threadIdx.x;
    const int wave    = tid >> 5;
    const int lane    = tid & 31;
    const int n16     = lane & 15;      // col within tile / A-row within tile
    const int half    = lane >> 4;      // C layout: 0 -> rows 0-7, 1 -> rows 8-15
    const int rtile   = wave & 3;       // row tile within block
    const int cpart   = wave >> 2;      // column partition
    const int rowbase = blockIdx.x * ROWS_BLK + rtile * 16;

    __shared__ float cs[32][CANDR];
    __shared__ int   ci[32][CANDR];
    __shared__ float wgt[ROWS_BLK][TOPK];
    __shared__ int   wid[ROWS_BLK][TOPK];

    const int ksubA = half * 16;   // A: halves split K 0..15 / 16..31 (same as B)
    const int ksubB = half * 16;   // B: halves split K 0..15 / 16..31

    // --- preload all 16 A fragments for this wave's row tile (128 VGPRs) ---
    v16h afrag[KSTEPS];
    {
        const _Float16* aptr = qn + (size_t)(rowbase + n16) * C_DIM;
        #pragma unroll
        for (int ks = 0; ks < KSTEPS; ++ks) {
            v8h alo = *(const v8h*)(aptr + ks * 32 + ksubA);
            v8h ahi = *(const v8h*)(aptr + ks * 32 + 8 + ksubA);
            afrag[ks] = __builtin_shufflevector(alo, ahi,
                           0, 1, 2, 3, 4, 5, 6, 7, 8, 9, 10, 11, 12, 13, 14, 15);
        }
    }

    // per-lane streaming top-5 for the 8 rows this lane sees in the C matrix
    float ts[8][TOPK];
    int   ti[8][TOPK];
    #pragma unroll
    for (int r = 0; r < 8; ++r)
        #pragma unroll
        for (int j = 0; j < TOPK; ++j) { ts[r][j] = -1e30f; ti[r][j] = 0; }

    const int ntiles = M >> 4;
    const int ntBeg  = cpart * (ntiles / CPARTS);
    const int ntEnd  = ntBeg + (ntiles / CPARTS);

    // issue all 16 B-fragment loads of one tile as a clause
    auto loadTile = [&](int nt, BTile& bt) {
        const _Float16* bptr = mn + (size_t)((nt << 4) + n16) * C_DIM + ksubB;
        #pragma unroll
        for (int ks = 0; ks < KSTEPS; ++ks)
            bt.f[ks] = *(const v16h*)(bptr + (size_t)ks * 32);
    };

    // 16 chained WMMAs + streaming top-5 update for one tile
    auto procTile = [&](int nt, const BTile& bt) {
        v8f acc = {};
        #pragma unroll
        for (int ks = 0; ks < KSTEPS; ++ks)
            acc = __builtin_amdgcn_wmma_f32_16x16x32_f16(
                      false, afrag[ks], false, bt.f[ks], (short)0, acc, false, false);
        const int col = (nt << 4) + n16;
        #pragma unroll
        for (int r = 0; r < 8; ++r) {
            float v = acc[r];
            if (v > ts[r][TOPK - 1]) {              // rare: whole-wave EXECZ skip
                float cv = v; int cid = col;
                #pragma unroll
                for (int j = 0; j < TOPK; ++j) {
                    bool gt  = cv > ts[r][j];
                    float os = ts[r][j]; int oi = ti[r][j];
                    ts[r][j] = gt ? cv : os;
                    ti[r][j] = gt ? cid : oi;
                    cv  = gt ? os : cv;
                    cid = gt ? oi : cid;
                }
            }
        }
    };

    // --- software-pipelined main loop: load tile n+1/n+2 under tile n/n+1 ---
    BTile bf0, bf1;
    if (ntBeg < ntEnd) loadTile(ntBeg, bf0);
    for (int nt = ntBeg; nt < ntEnd; nt += 2) {
        const bool has1 = (nt + 1) < ntEnd;
        if (has1) loadTile(nt + 1, bf1);            // prefetch into bf1
        __builtin_amdgcn_sched_barrier(0);          // pin loads above the WMMAs
        procTile(nt, bf0);
        if (nt + 2 < ntEnd) loadTile(nt + 2, bf0);  // prefetch into bf0
        __builtin_amdgcn_sched_barrier(0);
        if (has1) procTile(nt + 1, bf1);
    }

    // --- merge in two 32-row chunks (row tiles {0,1} then {2,3}) ---
    #pragma unroll
    for (int chunk = 0; chunk < 2; ++chunk) {
        if ((rtile >> 1) == chunk) {
            const int rl0  = (rtile & 1) * 16 + half * 8;  // local row in chunk
            const int slot = cpart * (16 * TOPK) + n16 * TOPK;
            #pragma unroll
            for (int r = 0; r < 8; ++r)
                #pragma unroll
                for (int j = 0; j < TOPK; ++j) {
                    cs[rl0 + r][slot + j] = ts[r][j];
                    ci[rl0 + r][slot + j] = ti[r][j];
                }
        }
        __syncthreads();

        if (tid < 32) {
            float ms[TOPK]; int mi[TOPK];
            #pragma unroll
            for (int j = 0; j < TOPK; ++j) { ms[j] = -1e30f; mi[j] = 0; }
            for (int i = 0; i < CANDR; ++i) {
                float cv = cs[tid][i]; int cid = ci[tid][i];
                if (cv > ms[TOPK - 1]) {
                    #pragma unroll
                    for (int j = 0; j < TOPK; ++j) {
                        bool gt  = cv > ms[j];
                        float os = ms[j]; int oi = mi[j];
                        ms[j] = gt ? cv : os;
                        mi[j] = gt ? cid : oi;
                        cv  = gt ? os : cv;
                        cid = gt ? oi : cid;
                    }
                }
            }
            const float mx = ms[0];              // sorted descending
            float e[TOPK], sum = 0.0f;
            #pragma unroll
            for (int j = 0; j < TOPK; ++j) { e[j] = __expf(ms[j] - mx); sum += e[j]; }
            const float inv = 1.0f / sum;
            const int lr = chunk * 32 + tid;
            #pragma unroll
            for (int j = 0; j < TOPK; ++j) { wgt[lr][j] = e[j] * inv; wid[lr][j] = mi[j]; }
        }
        __syncthreads();
    }

    // --- cooperative weighted gather from fp32 table + residual add ---
    const size_t blkrow0 = (size_t)blockIdx.x * ROWS_BLK;
    for (int e = tid; e < ROWS_BLK * C_DIM; e += THREADS) {
        const int lr = e >> 9;            // /512
        const int c  = e & (C_DIM - 1);
        float acc = 0.0f;
        #pragma unroll
        for (int j = 0; j < TOPK; ++j)
            acc += wgt[lr][j] * mem[(size_t)wid[lr][j] * C_DIM + c];
        const size_t g = (blkrow0 + lr) * C_DIM + c;
        out[g] = x[g] + 0.5f * acc;
    }
}

// ---------------------------------------------------------------------------
extern "C" void kernel_launch(void* const* d_in, const int* in_sizes, int n_in,
                              void* d_out, int out_size, void* d_ws, size_t ws_size,
                              hipStream_t stream) {
    const float* x   = (const float*)d_in[0];   // [B,S,C] fp32
    const float* mem = (const float*)d_in[1];   // [M,C]  fp32
    float* out = (float*)d_out;

    const int BS = in_sizes[0] / C_DIM;         // 8192
    const int M  = in_sizes[1] / C_DIM;         // 32768

    // workspace: [mn fp16: M*512][qn fp16: BS*512]  (~40 MB)
    _Float16* mn = (_Float16*)d_ws;
    _Float16* qn = (_Float16*)((char*)d_ws + (size_t)M * C_DIM * sizeof(_Float16));

    GPM_rownorm_f16<<<M,  128, 0, stream>>>(mem, mn);
    GPM_rownorm_f16<<<BS, 128, 0, stream>>>(x,   qn);
    GPM_simtopk<<<BS / ROWS_BLK, THREADS, 0, stream>>>(qn, mn, x, mem, out, M);
}
